// TypeSpecificGATLayer_47528108097936
// MI455X (gfx1250) — matrix operand order repaired and checked by reading, not compiled, not run on hardware
//
#include <hip/hip_runtime.h>

#define NN 4096
#define DD 256
#define SLOPE 0.2f

typedef __attribute__((ext_vector_type(16))) __bf16 v16bf;
typedef __attribute__((ext_vector_type(8)))  float  v8f;
typedef __attribute__((ext_vector_type(4)))  float    f32x4;
typedef __attribute__((ext_vector_type(4)))  unsigned u32x4;
typedef __attribute__((ext_vector_type(4)))  int      i32x4;

union Frag {
    v16bf v;
    unsigned u[8];
};

// ---- explicit global (addrspace 1) accessors: force global_load/store, not flat ----
typedef const f32x4  __attribute__((address_space(1)))* g_f4p;
typedef const u32x4  __attribute__((address_space(1)))* g_u4p;
typedef const float  __attribute__((address_space(1)))* g_fp;
typedef float        __attribute__((address_space(1)))* g_fpw;
typedef const __bf16 __attribute__((address_space(1)))* g_hp;
typedef __bf16       __attribute__((address_space(1)))* g_hpw;

__device__ __forceinline__ f32x4  gload4f(const float* p)  { return *(g_f4p)(unsigned long long)p; }
__device__ __forceinline__ u32x4  gload4u(const void* p)   { return *(g_u4p)(unsigned long long)p; }
__device__ __forceinline__ float  gloadf (const float* p)  { return *(g_fp )(unsigned long long)p; }
__device__ __forceinline__ __bf16 gloadh (const __bf16* p) { return *(g_hp )(unsigned long long)p; }
__device__ __forceinline__ void   gstoref(float* p, float v)   { *(g_fpw)(unsigned long long)p = v; }
__device__ __forceinline__ void   gstoreh(__bf16* p, __bf16 v) { *(g_hpw)(unsigned long long)p = v; }

__device__ __forceinline__ v8f wmma_bf16(v16bf a, v16bf b, v8f c) {
    return __builtin_amdgcn_wmma_f32_16x16x32_bf16(
        false, a, false, b, (short)0, c, false, false);
}

// async copy: per-lane 16B global -> LDS, tracked by ASYNCcnt
__device__ __forceinline__ void async_b128(void* lds, const void* gbase, unsigned voff) {
    unsigned ldsoff = (unsigned)(unsigned long)
        ( (__attribute__((address_space(3))) char*) lds );
    asm volatile("global_load_async_to_lds_b128 %0, %1, %2"
                 :: "v"(ldsoff), "v"(voff), "s"(gbase) : "memory");
}

// ---------------------------------------------------------------------------
// Kernel 1: Wh_r = H @ W_r^T (bf16 out). Block = 64 rows x 256 cols, 8 waves.
// ---------------------------------------------------------------------------
__global__ __launch_bounds__(256) void proj_kernel(
    const float* __restrict__ H,
    const float* __restrict__ W1, const float* __restrict__ W2,
    const float* __restrict__ W3,
    __bf16* __restrict__ Whb)
{
    const int r = blockIdx.z;
    const float* W = (r == 0) ? W1 : (r == 1) ? W2 : W3;
    __bf16* Wh = Whb + (size_t)r * NN * DD;

    const int tid  = threadIdx.x;
    const int wv   = tid >> 5;
    const int lane = tid & 31;
    const int rs   = wv & 3;
    const int ch   = wv >> 2;
    const int m    = lane & 15;
    const int g    = lane >> 4;

    const int rowbase = blockIdx.x * 64;
    const int arow    = rowbase + rs * 16 + m;

    const v8f vzero = {0.f,0.f,0.f,0.f,0.f,0.f,0.f,0.f};
    v8f acc[8];
    for (int t = 0; t < 8; t++) acc[t] = vzero;

    for (int kb = 0; kb < DD; kb += 32) {
        Frag af;
        {
            const float* hp = H + (size_t)arow * DD + kb + 8 * g;
            f32x4 h0 = gload4f(hp);
            f32x4 h1 = gload4f(hp + 4);
            f32x4 h2 = gload4f(hp + 16);
            f32x4 h3 = gload4f(hp + 20);
            af.v[0]  = (__bf16)h0.x; af.v[1]  = (__bf16)h0.y;
            af.v[2]  = (__bf16)h0.z; af.v[3]  = (__bf16)h0.w;
            af.v[4]  = (__bf16)h1.x; af.v[5]  = (__bf16)h1.y;
            af.v[6]  = (__bf16)h1.z; af.v[7]  = (__bf16)h1.w;
            af.v[8]  = (__bf16)h2.x; af.v[9]  = (__bf16)h2.y;
            af.v[10] = (__bf16)h2.z; af.v[11] = (__bf16)h2.w;
            af.v[12] = (__bf16)h3.x; af.v[13] = (__bf16)h3.y;
            af.v[14] = (__bf16)h3.z; af.v[15] = (__bf16)h3.w;
        }
        for (int t = 0; t < 8; t++) {
            const int col = ch * 128 + t * 16 + m;
            const float* wp = W + (size_t)col * DD + kb + 8 * g;
            f32x4 w0 = gload4f(wp);
            f32x4 w1 = gload4f(wp + 4);
            f32x4 w2 = gload4f(wp + 16);
            f32x4 w3 = gload4f(wp + 20);
            Frag bf;
            bf.v[0]  = (__bf16)w0.x; bf.v[1]  = (__bf16)w0.y;
            bf.v[2]  = (__bf16)w0.z; bf.v[3]  = (__bf16)w0.w;
            bf.v[4]  = (__bf16)w1.x; bf.v[5]  = (__bf16)w1.y;
            bf.v[6]  = (__bf16)w1.z; bf.v[7]  = (__bf16)w1.w;
            bf.v[8]  = (__bf16)w2.x; bf.v[9]  = (__bf16)w2.y;
            bf.v[10] = (__bf16)w2.z; bf.v[11] = (__bf16)w2.w;
            bf.v[12] = (__bf16)w3.x; bf.v[13] = (__bf16)w3.y;
            bf.v[14] = (__bf16)w3.z; bf.v[15] = (__bf16)w3.w;
            acc[t] = wmma_bf16(af.v, bf.v, acc[t]);
        }
    }

    for (int t = 0; t < 8; t++) {
        const int col = ch * 128 + t * 16 + m;
        for (int v = 0; v < 8; v++) {
            const int rr = rowbase + rs * 16 + v + 8 * g;
            gstoreh(Wh + (size_t)rr * DD + col, (__bf16)acc[t][v]);
        }
    }
}

// ---------------------------------------------------------------------------
// Kernel 2: src_i = Wh_i . a[:256], dst_i = Wh_i . a[256:]  (wave per row)
// ---------------------------------------------------------------------------
__global__ __launch_bounds__(256) void srcdst_kernel(
    const __bf16* __restrict__ Whb,
    const float* __restrict__ a1, const float* __restrict__ a2,
    const float* __restrict__ a3,
    float* __restrict__ src, float* __restrict__ dst)
{
    const int r = blockIdx.z;
    const float* a = (r == 0) ? a1 : (r == 1) ? a2 : a3;
    const __bf16* Wh = Whb + (size_t)r * NN * DD;

    const int wv   = threadIdx.x >> 5;
    const int lane = threadIdx.x & 31;
    const int i    = blockIdx.x * 8 + wv;

    float ss = 0.f, ds = 0.f;
    for (int c = lane; c < DD; c += 32) {
        float w = (float)gloadh(Wh + (size_t)i * DD + c);
        ss += w * gloadf(a + c);
        ds += w * gloadf(a + DD + c);
    }
    for (int off = 16; off; off >>= 1) {
        ss += __shfl_xor(ss, off);
        ds += __shfl_xor(ds, off);
    }
    if (lane == 0) {
        gstoref(src + r * NN + i, ss);
        gstoref(dst + r * NN + i, ds);
    }
}

// ---------------------------------------------------------------------------
// Kernel 3: maxd[r] = max_j dst_r[j]
// ---------------------------------------------------------------------------
__global__ __launch_bounds__(256) void maxd_kernel(
    const float* __restrict__ dst, float* __restrict__ maxd)
{
    const int r = blockIdx.x;
    __shared__ float red[256];
    float mx = -3.0e38f;
    for (int i = threadIdx.x; i < NN; i += 256)
        mx = fmaxf(mx, gloadf(dst + r * NN + i));
    red[threadIdx.x] = mx;
    __syncthreads();
    for (int s = 128; s; s >>= 1) {
        if (threadIdx.x < s)
            red[threadIdx.x] = fmaxf(red[threadIdx.x], red[threadIdx.x + s]);
        __syncthreads();
    }
    if (threadIdx.x == 0) gstoref(maxd + r, red[0]);
}

// ---------------------------------------------------------------------------
// Kernel 4: fused masked-softmax + alpha@Wh, all 3 relations.
// Block = 64 rows x 128 cols. A-tiles double-buffered via async LDS copies.
// ---------------------------------------------------------------------------
__global__ __launch_bounds__(256) void agg_kernel(
    const int* __restrict__ A1, const int* __restrict__ A2,
    const int* __restrict__ A3,
    const __bf16* __restrict__ Whb,
    const float* __restrict__ src, const float* __restrict__ dst,
    const float* __restrict__ maxd, const float* __restrict__ bias,
    float* __restrict__ out)
{
    __shared__ int            s_A[2][64][32];     // adjacency tiles (16 KB)
    __shared__ __bf16         s_P[64][32];        // attention tile (4 KB)
    __shared__ unsigned short s_Wt[128][34];      // Wh panel, K-transposed, padded
    __shared__ float          s_s[64], s_c[64], s_l[64];

    const int tid  = threadIdx.x;
    const int wv   = tid >> 5;
    const int lane = tid & 31;
    const int rs   = wv & 3;      // row strip
    const int cq   = wv >> 2;     // column quarter (64 cols)
    const int m    = lane & 15;
    const int g    = lane >> 4;

    const int rowbase = blockIdx.x * 64;
    const int colblk  = blockIdx.y * 128;

    const v8f vzero = {0.f,0.f,0.f,0.f,0.f,0.f,0.f,0.f};
    v8f carry[4] = {vzero, vzero, vzero, vzero};

    const int prow = tid >> 2;        // P-tile row this thread fills
    const int pj0  = (tid & 3) * 8;   // 8 consecutive P columns

    // chunk geometry for async A-tile fill: 512 x 16B chunks, 2 per thread
    const int c0row = tid >> 3,         c0seg = tid & 7;
    const int c1row = (tid + 256) >> 3, c1seg = (tid + 256) & 7;

    // staging geometry for Wh panel: 32 rows x 128 cols, 16 bf16 per thread
    const int wrow = tid >> 3;
    const int wseg = (tid & 7) * 16;

    for (int r = 0; r < 3; r++) {
        const int*    A  = (r == 0) ? A1 : (r == 1) ? A2 : A3;
        const float*  sr = src + r * NN;
        const float*  dr = dst + r * NN;
        const __bf16* Wh = Whb + (size_t)r * NN * DD;

        if (tid < 64) {
            float sv = gloadf(sr + rowbase + tid);
            s_s[tid] = sv;
            float cc = sv + gloadf(maxd + r);
            s_c[tid] = (cc > 0.f) ? cc : SLOPE * cc;
            s_l[tid] = 0.f;
        }

        // prime A-tile buffer 0 (jb = 0)
        {
            const char* gb = (const char*)(A + (size_t)rowbase * NN);
            async_b128(&s_A[0][c0row][c0seg * 4], gb,
                       (unsigned)(c0row * (NN * 4) + c0seg * 16));
            async_b128(&s_A[0][c1row][c1seg * 4], gb,
                       (unsigned)(c1row * (NN * 4) + c1seg * 16));
        }
        __syncthreads();

        v8f acc[4] = {vzero, vzero, vzero, vzero};

        for (int jb = 0; jb < NN; jb += 32) {
            const int cur = (jb >> 5) & 1;
            const int nxt = cur ^ 1;

            // issue async fill of next A tile, then wait for current
            if (jb + 32 < NN) {
                const char* gb = (const char*)(A + (size_t)rowbase * NN + (jb + 32));
                async_b128(&s_A[nxt][c0row][c0seg * 4], gb,
                           (unsigned)(c0row * (NN * 4) + c0seg * 16));
                async_b128(&s_A[nxt][c1row][c1seg * 4], gb,
                           (unsigned)(c1row * (NN * 4) + c1seg * 16));
                asm volatile("s_wait_asynccnt 2" ::: "memory");
            } else {
                asm volatile("s_wait_asynccnt 0" ::: "memory");
            }

            // stage Wh panel [jb..jb+31][colblk..colblk+127] into LDS transposed
            {
                const __bf16* wp = Wh + (size_t)(jb + wrow) * DD + colblk + wseg;
                u32x4 q0 = gload4u(wp);
                u32x4 q1 = gload4u(wp + 8);
                unsigned qq[8] = {q0.x, q0.y, q0.z, q0.w, q1.x, q1.y, q1.z, q1.w};
                for (int j = 0; j < 8; j++) {
                    s_Wt[wseg + 2 * j][wrow]     = (unsigned short)(qq[j] & 0xffffu);
                    s_Wt[wseg + 2 * j + 1][wrow] = (unsigned short)(qq[j] >> 16);
                }
            }
            __syncthreads();   // A tile (async) + Wh panel ready

            // ---- build P tile (64x32) from LDS A tile ----
            const i32x4* ap = (const i32x4*)&s_A[cur][prow][pj0];
            i32x4 av0 = ap[0];
            i32x4 av1 = ap[1];
            f32x4 d0 = gload4f(dr + jb + pj0);
            f32x4 d1 = gload4f(dr + jb + pj0 + 4);

            const float sv = s_s[prow];
            const float cv = s_c[prow];
            int   avv[8] = {av0.x, av0.y, av0.z, av0.w, av1.x, av1.y, av1.z, av1.w};
            float dvv[8] = {d0.x, d0.y, d0.z, d0.w, d1.x, d1.y, d1.z, d1.w};
            float psum = 0.f;
            for (int jj = 0; jj < 8; jj++) {
                float e = sv + dvv[jj];
                e = (e > 0.f) ? e : SLOPE * e;
                float p = (avv[jj] != 0) ? __expf(e - cv) : 0.f;
                psum += p;
                s_P[prow][pj0 + jj] = (__bf16)p;
            }
            atomicAdd(&s_l[prow], psum);    // ds_add_f32
            __syncthreads();

            // ---- A fragment from P tile (ISA 16-bit A layout) ----
            Frag af;
            for (int v = 0; v < 8; v++) {
                const int k0 = ((v & 4) ? 16 : 0) + (v & 3) * 2 + 8 * g;
                af.u[v] = *(const unsigned*)&s_P[rs * 16 + m][k0];
            }

            // ---- B fragments from transposed LDS panel + WMMA ----
            for (int t = 0; t < 4; t++) {
                const int colL = cq * 64 + t * 16 + m;
                Frag bf;
                for (int v = 0; v < 8; v++)
                    bf.u[v] = *(const unsigned*)&s_Wt[colL][((v & 4) ? 16 : 0) + (v & 3) * 2 + 8 * g];
                acc[t] = wmma_bf16(af.v, bf.v, acc[t]);
            }
            __syncthreads();   // P / Wt reused next iteration
        }

        // ---- normalize rows by 1/l and fold into carry ----
        float sc[8];
        for (int v = 0; v < 8; v++) {
            float l = s_l[rs * 16 + v + 8 * g];
            sc[v] = (l > 0.f) ? (1.f / l) : 0.f;
        }
        for (int t = 0; t < 4; t++)
            for (int v = 0; v < 8; v++)
                carry[t][v] += acc[t][v] * sc[v];
        __syncthreads();
    }

    // ---- single output store, bias folded in ----
    for (int t = 0; t < 4; t++) {
        const int col = colblk + cq * 64 + t * 16 + m;
        const float b = gloadf(bias + col);
        for (int v = 0; v < 8; v++) {
            const int row = rowbase + rs * 16 + v + 8 * g;
            gstoref(out + (size_t)row * DD + col, carry[t][v] + b);
        }
    }
}

// ---------------------------------------------------------------------------
extern "C" void kernel_launch(void* const* d_in, const int* in_sizes, int n_in,
                              void* d_out, int out_size, void* d_ws, size_t ws_size,
                              hipStream_t stream)
{
    const float* H    = (const float*)d_in[0];
    const float* W1   = (const float*)d_in[1];
    const float* W2   = (const float*)d_in[2];
    const float* W3   = (const float*)d_in[3];
    const float* a1   = (const float*)d_in[4];
    const float* a2   = (const float*)d_in[5];
    const float* a3   = (const float*)d_in[6];
    const float* bias = (const float*)d_in[7];
    const int*   A1   = (const int*)d_in[8];
    const int*   A2   = (const int*)d_in[9];
    const int*   A3   = (const int*)d_in[10];
    float* out = (float*)d_out;

    char* ws = (char*)d_ws;
    __bf16* Whb = (__bf16*)ws;
    size_t off = (size_t)3 * NN * DD * sizeof(__bf16);
    float* src  = (float*)(ws + off); off += (size_t)3 * NN * sizeof(float);
    float* dst  = (float*)(ws + off); off += (size_t)3 * NN * sizeof(float);
    float* maxd = (float*)(ws + off);

    proj_kernel<<<dim3(NN / 64, 1, 3), 256, 0, stream>>>(H, W1, W2, W3, Whb);
    srcdst_kernel<<<dim3(NN / 8, 1, 3), 256, 0, stream>>>(Whb, a1, a2, a3, src, dst);
    maxd_kernel<<<dim3(3), 256, 0, stream>>>(dst, maxd);
    agg_kernel<<<dim3(NN / 64, 2), 256, 0, stream>>>(A1, A2, A3, Whb, src, dst,
                                                     maxd, bias, out);
}
